// GCN_6536940224662
// MI455X (gfx1250) — matrix operand order repaired and checked by reading, not compiled, not run on hardware
//
#include <hip/hip_runtime.h>
#include <hip/hip_bf16.h>
#include <math.h>

// GCN forward for MI455X (gfx1250, wave32).
// Dense GEMMs -> v_wmma_f32_16x16x32_bf16 (bf16 operands, fp32 accum),
// A tile staged in LDS (conflict-free padded layout), 4 col-tiles per wave
// for A-fragment reuse. Edge gather/scatter -> fp32, L2-resident
// (38MB matrices << 192MB L2), hardware global_atomic_add_f32 segment-sum.

#define HID 192

typedef __bf16 bf16_t;
typedef __attribute__((ext_vector_type(16))) __bf16 v16bf;
typedef __attribute__((ext_vector_type(8)))  __bf16 v8bf;
typedef __attribute__((ext_vector_type(8)))  float  v8f;

__device__ __forceinline__ void atomAddF(float* p, float v) {
  __hip_atomic_fetch_add(p, v, __ATOMIC_RELAXED, __HIP_MEMORY_SCOPE_AGENT);
}

// ---------------- small utility kernels ----------------
__global__ void k_fill_f32(float* __restrict__ p, float v, long n) {
  long t = (long)blockIdx.x * blockDim.x + threadIdx.x;
  if (t < n) p[t] = v;
}

__global__ void k_deg_edges(const int* __restrict__ dst, float* __restrict__ deg, int E) {
  int t = blockIdx.x * blockDim.x + threadIdx.x;
  if (t < E) atomAddF(&deg[dst[t]], 1.0f);
}

__global__ void k_rsqrt_inplace(float* __restrict__ d, int n) {
  int t = blockIdx.x * blockDim.x + threadIdx.x;
  if (t < n) d[t] = rsqrtf(fmaxf(d[t], 1.0f));
}

// f32 -> bf16, optional per-row scale (folds dinv into the GEMM A operand)
__global__ void k_cvt_bf16(const float* __restrict__ in, bf16_t* __restrict__ out,
                           const float* __restrict__ rowscale, int cols, long total) {
  long t = (long)blockIdx.x * blockDim.x + threadIdx.x;
  if (t >= total) return;
  long r = t / cols;
  float s = rowscale ? rowscale[r] : 1.0f;
  out[t] = (bf16_t)(in[t] * s);
}

// W[Kin][Kout] f32 row-major  ->  Wt[Kout][Kin] bf16 (contiguous-K per column)
__global__ void k_wt_cvt(const float* __restrict__ W, bf16_t* __restrict__ Wt,
                         int Kin, int Kout) {
  int t = blockIdx.x * blockDim.x + threadIdx.x;
  if (t >= Kin * Kout) return;
  int k = t / Kout, n = t - k * Kout;
  Wt[(size_t)n * Kin + k] = (bf16_t)W[t];
}

// ---------------- WMMA GEMM: C[N,Kout] = A[N,Kin] @ W + (bias) ----------------
// Block = 128 threads (4 waves). Block owns 16 rows; each wave owns 4 column
// tiles (16 cols each) => block covers up to 256 output columns, grid.y == 1.
// The 16 x KIN A tile is staged once into LDS (row stride KIN+8 so the
// 16-lane ds_load_b128 fragment reads hit distinct banks), then each wave
// reuses one A fragment across 4 WMMAs per K-step.
//
// A fragment (16-bit A, ISA 7.12.2): lane<16 holds row M=lane,
// K = k0+[0..7] (elems 0-7) and k0+[16..23] (elems 8-15); lanes 16-31 same
// rows, K offsets +8/+24.
// B fragment: lane<16 holds column N=n0+lane, same K order as A (k0+[0..7],
// k0+[16..23]); lanes 16-31 K offsets +8/+24 — from the pre-transposed Wt.
template <int KIN>
__global__ __launch_bounds__(128)
void k_wmma_gemm(const bf16_t* __restrict__ A, const bf16_t* __restrict__ Wt,
                 const float* __restrict__ bias, float* __restrict__ C,
                 int Kout, int doRelu) {
  constexpr int LDK = KIN + 8;                 // padded LDS row stride (bank-spread)
  __shared__ __align__(16) bf16_t As[16 * LDK];

  const int m0 = blockIdx.x * 16;

  // cooperative stage: 16 rows x KIN bf16 -> LDS (16B per thread per step)
  constexpr int CHUNKS = 16 * KIN / 8;         // # of v8bf chunks
  for (int c = threadIdx.x; c < CHUNKS; c += 128) {
    int row  = c / (KIN / 8);
    int col8 = c - row * (KIN / 8);
    *(v8bf*)(&As[row * LDK + col8 * 8]) =
        *(const v8bf*)(A + (size_t)(m0 + row) * KIN + col8 * 8);
  }
  __syncthreads();

  const int lane = threadIdx.x & 31;
  const int wave = threadIdx.x >> 5;
  const int lh = lane & 15;
  const int hi = lane >> 4;

  const int nTiles = Kout >> 4;
  const int tile0 = wave * 4;                  // 4 consecutive col tiles per wave
  if (tile0 >= nTiles) return;                 // wave-uniform (EXEC stays all-ones)

  const bf16_t* as_frag = &As[lh * LDK + hi * 8];

  v8f acc[4];
#pragma unroll
  for (int t = 0; t < 4; ++t) acc[t] = v8f{0.f, 0.f, 0.f, 0.f, 0.f, 0.f, 0.f, 0.f};

#pragma unroll
  for (int k0 = 0; k0 < KIN; k0 += 32) {
    // A fragment from LDS (shared by all 4 tiles of this wave)
    v8bf a_lo = *(const v8bf*)(as_frag + k0);        // K k0+hi*8 + [0..7]
    v8bf a_hi = *(const v8bf*)(as_frag + k0 + 16);   // K k0+hi*8 + [16..23]
    v16bf a;
#pragma unroll
    for (int i = 0; i < 8; ++i) { a[i] = a_lo[i]; a[8 + i] = a_hi[i]; }

#pragma unroll
    for (int t = 0; t < 4; ++t) {
      if (tile0 + t < nTiles) {                      // wave-uniform per tile
        const bf16_t* bcol =
            Wt + (size_t)((tile0 + t) * 16 + lh) * KIN + hi * 8;
        v8bf b_lo = *(const v8bf*)(bcol + k0);       // K k0+hi*8 + [0..7]
        v8bf b_hi = *(const v8bf*)(bcol + k0 + 16);  // K k0+hi*8 + [16..23]
        v16bf b;
#pragma unroll
        for (int i = 0; i < 8; ++i) { b[i] = b_lo[i]; b[8 + i] = b_hi[i]; }
        acc[t] = __builtin_amdgcn_wmma_f32_16x16x32_bf16(
            /*neg_a=*/false, a, /*neg_b=*/false, b,
            /*c_mod=*/(short)0, acc[t], /*reuse_a=*/false, /*reuse_b=*/false);
      }
    }
  }

  // C/D layout: VGPR r -> row m0+r (lanes 0-15) / row m0+8+r (lanes 16-31),
  // col = n0 + lane%16. Bias is constant per column -> hoisted to one load.
#pragma unroll
  for (int t = 0; t < 4; ++t) {
    if (tile0 + t < nTiles) {
      const int n = (tile0 + t) * 16 + lh;
      const float bval = bias ? bias[n] : 0.0f;
      float* crow = C + (size_t)(m0 + hi * 8) * Kout + n;
#pragma unroll
      for (int r = 0; r < 8; ++r) {
        float v = acc[t][r] + bval;
        if (doRelu) v = fmaxf(v, 0.0f);
        crow[(size_t)r * Kout] = v;
      }
    }
  }
}

// ---------------- edge scatter-add: acc[dst] += msca[src] ----------------
// One thread = one float4 of one edge (48 consecutive threads cover a row ->
// fully coalesced gather); 4 global_atomic_add_f32 per thread into L2.
__global__ void k_scatter(const int* __restrict__ src, const int* __restrict__ dst,
                          const float4* __restrict__ msca, float* __restrict__ acc, int E) {
  long t = (long)blockIdx.x * blockDim.x + threadIdx.x;
  const int Q = HID / 4;
  long total = (long)E * Q;
  if (t >= total) return;
  int e = (int)(t / Q);
  int q = (int)(t - (long)e * Q);
  int s = src[e], d = dst[e];
  float4 v = msca[(size_t)s * Q + q];
  float* b = acc + (size_t)d * HID + q * 4;
  atomAddF(b + 0, v.x);
  atomAddF(b + 1, v.y);
  atomAddF(b + 2, v.z);
  atomAddF(b + 3, v.w);
}

// h += relu(dinv * (acc + self_loop_msg) + convB)
__global__ void k_conv_finalize(float* __restrict__ h, const float* __restrict__ acc,
                                const float* __restrict__ msca, const float* __restrict__ dinv,
                                const float* __restrict__ bias, long total) {
  long t = (long)blockIdx.x * blockDim.x + threadIdx.x;
  if (t >= total) return;
  long r = t / HID;
  int  c = (int)(t - r * HID);
  float agg = dinv[r] * (acc[t] + msca[t]) + bias[c];
  h[t] += fmaxf(agg, 0.0f);
}

// final tiny layers + sigmoid + unit-normalize, fused
__global__ void k_heads(const float* __restrict__ p2, const float* __restrict__ r1,
                        const float* __restrict__ Wp3, const float* __restrict__ bp3,
                        const float* __restrict__ Wr2, const float* __restrict__ br2,
                        float* __restrict__ out, int N) {
  int i = blockIdx.x * blockDim.x + threadIdx.x;
  if (i >= N) return;
  float px = bp3[0], py = bp3[1], rr = br2[0];
  const float* p = p2 + (size_t)i * 96;
  const float* q = r1 + (size_t)i * 96;
#pragma unroll 4
  for (int k = 0; k < 96; ++k) {
    float a = p[k];
    px += a * Wp3[2 * k];
    py += a * Wp3[2 * k + 1];
    rr += q[k] * Wr2[k];
  }
  float radius = 1.0f / (1.0f + __expf(-rr));
  float nrm = sqrtf(px * px + py * py) + 1e-8f;
  float inv = radius / nrm;
  out[2 * i]     = px * inv;
  out[2 * i + 1] = py * inv;
}

// ---------------- host-side orchestration ----------------
static inline unsigned gridFor(long n, int block) { return (unsigned)((n + block - 1) / block); }

extern "C" void kernel_launch(void* const* d_in, const int* in_sizes, int n_in,
                              void* d_out, int out_size, void* d_ws, size_t ws_size,
                              hipStream_t stream) {
  const float* x     = (const float*)d_in[0];
  const int*   ei    = (const int*)d_in[1];
  const float* Wp    = (const float*)d_in[2];
  const float* bp    = (const float*)d_in[3];
  const float* convW = (const float*)d_in[4];
  const float* convB = (const float*)d_in[5];
  const float* Wp1   = (const float*)d_in[6];
  const float* bp1   = (const float*)d_in[7];
  const float* Wp2   = (const float*)d_in[8];
  const float* bp2   = (const float*)d_in[9];
  const float* Wp3   = (const float*)d_in[10];
  const float* bp3   = (const float*)d_in[11];
  const float* Wr1   = (const float*)d_in[12];
  const float* br1   = (const float*)d_in[13];
  const float* Wr2   = (const float*)d_in[14];
  const float* br2   = (const float*)d_in[15];

  const int N = in_sizes[0] / 128;   // 50000 (divisible by 16)
  const int E = in_sizes[1] / 2;     // 800000
  const int* srcI = ei;
  const int* dstI = ei + E;

  // workspace carve-out (all offsets 256B-aligned)
  float* h    = (float*)d_ws;                    // N x 192 f32
  float* msca = h    + (size_t)N * HID;          // N x 192 f32 (m_scaled / p1)
  float* acc  = msca + (size_t)N * HID;          // N x 192 f32 (segment sum / p2+r1)
  float* dinv = acc  + (size_t)N * HID;          // N f32 (deg, then rsqrt in place)
  size_t nPad = ((size_t)N + 63) & ~(size_t)63;
  bf16_t* ab16  = (bf16_t*)(dinv + nPad);        // N x 192 bf16 staging (A operand)
  bf16_t* wpT   = ab16 + (size_t)N * HID;        // 192x128
  bf16_t* convT = wpT  + 192 * 128;              // 4 x 192x192
  bf16_t* wp1T  = convT + 4 * 192 * 192;         // 192x192
  bf16_t* wp2T  = wp1T + 192 * 192;              // 96x192
  bf16_t* wr1T  = wp2T + 96 * 192;               // 96x192

  const int BLK = 256;

  // 1) weights -> bf16 transposed
  k_wt_cvt<<<gridFor(128 * 192, BLK), BLK, 0, stream>>>(Wp, wpT, 128, 192);
  for (int i = 0; i < 4; ++i)
    k_wt_cvt<<<gridFor(192 * 192, BLK), BLK, 0, stream>>>(convW + (size_t)i * 192 * 192,
                                                          convT + (size_t)i * 192 * 192, 192, 192);
  k_wt_cvt<<<gridFor(192 * 192, BLK), BLK, 0, stream>>>(Wp1, wp1T, 192, 192);
  k_wt_cvt<<<gridFor(192 * 96, BLK), BLK, 0, stream>>>(Wp2, wp2T, 192, 96);
  k_wt_cvt<<<gridFor(192 * 96, BLK), BLK, 0, stream>>>(Wr1, wr1T, 192, 96);

  // 2) symmetric-norm factors: deg (with self loop) -> dinv = rsqrt(deg)
  k_fill_f32<<<gridFor(N, BLK), BLK, 0, stream>>>(dinv, 1.0f, N);
  k_deg_edges<<<gridFor(E, BLK), BLK, 0, stream>>>(dstI, dinv, E);
  k_rsqrt_inplace<<<gridFor(N, BLK), BLK, 0, stream>>>(dinv, N);

  // 3) input projection: h = x @ Wp + bp
  k_cvt_bf16<<<gridFor((long)N * 128, BLK), BLK, 0, stream>>>(x, ab16, (const float*)nullptr,
                                                              128, (long)N * 128);
  dim3 gemmBlk(128);
  dim3 gemmGrid(N / 16);  // block covers all <=256 output cols (grid.y == 1)
  k_wmma_gemm<128><<<gemmGrid, gemmBlk, 0, stream>>>(ab16, wpT, bp, h, 192, 0);

  // 4) four GCNConv layers
  for (int i = 0; i < 4; ++i) {
    // m_scaled = (dinv .* h) @ convW[i]  (dinv folded into bf16 conversion)
    k_cvt_bf16<<<gridFor((long)N * HID, BLK), BLK, 0, stream>>>(h, ab16, dinv, HID, (long)N * HID);
    k_wmma_gemm<192><<<gemmGrid, gemmBlk, 0, stream>>>(ab16, convT + (size_t)i * 192 * 192,
                                                       (const float*)nullptr, msca, 192, 0);
    hipMemsetAsync(acc, 0, (size_t)N * HID * sizeof(float), stream);
    k_scatter<<<gridFor((long)E * (HID / 4), BLK), BLK, 0, stream>>>(srcI, dstI,
                                                                     (const float4*)msca, acc, E);
    // h += relu(dinv * (acc + self_msg) + convB[i])
    k_conv_finalize<<<gridFor((long)N * HID, BLK), BLK, 0, stream>>>(h, acc, msca, dinv,
                                                                     convB + (size_t)i * HID,
                                                                     (long)N * HID);
  }

  // 5) heads
  float* p2 = acc;                   // N x 96
  float* r1 = acc + (size_t)N * 96;  // N x 96
  k_cvt_bf16<<<gridFor((long)N * HID, BLK), BLK, 0, stream>>>(h, ab16, (const float*)nullptr,
                                                              HID, (long)N * HID);
  k_wmma_gemm<192><<<gemmGrid, gemmBlk, 0, stream>>>(ab16, wp1T, bp1, msca, 192, 1);  // p1
  k_wmma_gemm<192><<<gemmGrid, gemmBlk, 0, stream>>>(ab16, wr1T, br1, r1, 96, 1);     // r1 (from h)
  k_cvt_bf16<<<gridFor((long)N * HID, BLK), BLK, 0, stream>>>(msca, ab16, (const float*)nullptr,
                                                              HID, (long)N * HID);
  k_wmma_gemm<192><<<gemmGrid, gemmBlk, 0, stream>>>(ab16, wp2T, bp2, p2, 96, 1);     // p2 (from p1)

  // 6) tiny final layers + sigmoid + normalize
  k_heads<<<gridFor(N, BLK), BLK, 0, stream>>>(p2, r1, Wp3, bp3, Wr2, br2, (float*)d_out, N);
}
